// AdvancedGraphNeuralNetwork_2920577761625
// MI455X (gfx1250) — matrix operand order repaired and checked by reading, not compiled, not run on hardware
//
#include <hip/hip_runtime.h>

#define N_NODES 20000
#define N_EDGES 200000
#define F_INPUT 20
#define HID 128
#define HEADS 4
#define NLAYERS 4
#define NGRAPH 32
#define NEG_SLOPE 0.2f
#define ETOT (N_EDGES + N_NODES)
#define NC (HEADS * HID) /* 512 */

typedef __attribute__((ext_vector_type(16))) _Float16 v16h;
typedef __attribute__((ext_vector_type(8)))  float    v8f;
typedef __attribute__((ext_vector_type(4)))  unsigned int v4u;

union Frag16 { v16h h; v4u u[2]; };

__device__ __forceinline__ float leakyf(float x) { return x > 0.f ? x : NEG_SLOPE * x; }

__device__ __forceinline__ void atomicMaxF(float* a, float v) {
  // sign-split trick: monotone under int max for >=0, uint min for <0
  if (v >= 0.f) atomicMax((int*)a, __float_as_int(v));
  else          atomicMin((unsigned int*)a, __float_as_uint(v));
}

// ---------------- utility fills / converts ----------------
__global__ void fill_f32(float* __restrict__ p, float v, int n) {
  int i = blockIdx.x * blockDim.x + threadIdx.x;
  if (i < n) p[i] = v;
}

__global__ void cvt_f16(const float* __restrict__ s, _Float16* __restrict__ d, int n) {
  int i = blockIdx.x * blockDim.x + threadIdx.x;
  if (i < n) d[i] = (_Float16)s[i];
}

// Wt[n*HID + k] = (f16) gat_W_l[k*NC + n]   (transpose so B columns are k-contiguous)
__global__ void wt_f16(const float* __restrict__ gw, _Float16* __restrict__ wt) {
  int t = blockIdx.x * blockDim.x + threadIdx.x;
  if (t >= NC * HID) return;
  int n = t / HID, k = t % HID;
  wt[t] = (_Float16)gw[(size_t)k * NC + n];
}

// ---------------- input projection: h = relu(x @ W_in + b_in) ----------------
__global__ void proj_relu(const float* __restrict__ x, const float* __restrict__ W,
                          const float* __restrict__ b, float* __restrict__ h) {
  int t = blockIdx.x * blockDim.x + threadIdx.x;
  if (t >= N_NODES * HID) return;
  int n = t / HID, c = t % HID;
  float acc = b[c];
#pragma unroll
  for (int k = 0; k < F_INPUT; ++k) acc += x[n * F_INPUT + k] * W[k * HID + c];
  h[t] = fmaxf(acc, 0.f);
}

// ---------------- WMMA GEMM: C[M,NC] = A[M,K] * Bt[NC,K]^T  (f16 in, f32 out) ----------------
// Register-blocked: each wave computes a 16(M) x 64(N) tile -> 4 accumulators share one
// A fragment per k-step (4x less A gather traffic, 16 v_wmma per wave).
__global__ __launch_bounds__(256) void gemm_wmma_f16(const _Float16* __restrict__ A,
                                                     const _Float16* __restrict__ Bt,
                                                     float* __restrict__ C) {
  const int K = HID;
  const int ngroups = NC / 64;                     // 8 groups of four 16-wide N tiles
  int wid = blockIdx.x * 8 + (threadIdx.x >> 5);   // 8 waves per block
  int mt  = wid / ngroups;
  int ng  = wid % ngroups;
  if (mt >= N_NODES / 16) return;                  // wave-uniform: EXEC stays all-ones

  int lane  = threadIdx.x & 31;
  int l15   = lane & 15;
  int lhalf = lane >> 4;                           // 0: lanes 0-15, 1: lanes 16-31
  int arow  = mt * 16 + l15;                       // A: lane holds matrix row M
  int akoff = lhalf * 8;                           // A halves: K {0..7,16..23} vs {8..15,24..31}
  int bkoff = lhalf * 8;                           // B halves: K {0..7,16..23} vs {8..15,24..31}

  const _Float16* ap = A + (size_t)arow * K;
  const _Float16* bp = Bt + (size_t)(ng * 64 + l15) * K;  // sub-tile s at +s*16*K

  v8f acc[4] = {v8f{}, v8f{}, v8f{}, v8f{}};
#pragma unroll
  for (int ks = 0; ks < K; ks += 32) {
    Frag16 fa;
    fa.u[0] = *(const v4u*)(ap + ks + akoff);        // 8 halves: K = ks+akoff .. +7
    fa.u[1] = *(const v4u*)(ap + ks + akoff + 16);   // 8 halves: K = ks+akoff+16 .. +23
#pragma unroll
    for (int s = 0; s < 4; ++s) {
      Frag16 fb;
      const _Float16* bs = bp + (size_t)s * 16 * K;
      fb.u[0] = *(const v4u*)(bs + ks + bkoff);      // 8 halves: K = ks+bkoff .. +7
      fb.u[1] = *(const v4u*)(bs + ks + bkoff + 16); // 8 halves: K = ks+bkoff+16 .. +23
      acc[s] = __builtin_amdgcn_wmma_f32_16x16x32_f16(
          /*neg_a=*/false, fa.h, /*neg_b=*/false, fb.h,
          /*c_mod=*/(short)0, acc[s], /*reuse_a=*/false, /*reuse_b=*/false);
    }
  }

#pragma unroll
  for (int s = 0; s < 4; ++s) {
    int ncol = ng * 64 + s * 16 + l15;
#pragma unroll
    for (int r = 0; r < 8; ++r) {                  // C/D: vgpr r -> M = r or 8+r
      int mrow = mt * 16 + lhalf * 8 + r;
      C[(size_t)mrow * NC + ncol] = acc[s][r];
    }
  }
}

// ---------------- attention logits: al_src/al_dst [N, HEADS] ----------------
__global__ void att_dots(const float* __restrict__ xl, const float* __restrict__ as,
                         const float* __restrict__ ad, float* __restrict__ als,
                         float* __restrict__ ald) {
  int t = blockIdx.x * blockDim.x + threadIdx.x;
  if (t >= N_NODES * HEADS) return;
  int n = t / HEADS, h = t % HEADS;
  const float* row = xl + (size_t)n * NC + h * HID;
  const float* vs = as + h * HID;
  const float* vd = ad + h * HID;
  float s = 0.f, d = 0.f;
#pragma unroll 4
  for (int c = 0; c < HID; ++c) { float v = row[c]; s += v * vs[c]; d += v * vd[c]; }
  als[t] = s;
  ald[t] = d;
}

// ---------------- edge passes (self-loops appended: e >= N_EDGES -> src=dst=e-N_EDGES) ----------------
__global__ void edge_max(const int* __restrict__ ei, const float* __restrict__ als,
                         const float* __restrict__ ald, float* __restrict__ m) {
  int e = blockIdx.x * blockDim.x + threadIdx.x;
  if (e >= ETOT) return;
  int s = (e < N_EDGES) ? ei[e] : (e - N_EDGES);
  int d = (e < N_EDGES) ? ei[N_EDGES + e] : (e - N_EDGES);
#pragma unroll
  for (int h = 0; h < HEADS; ++h) {
    float v = leakyf(als[s * HEADS + h] + ald[d * HEADS + h]);
    atomicMaxF(&m[d * HEADS + h], v);
  }
}

__global__ void edge_expsum(const int* __restrict__ ei, const float* __restrict__ als,
                            const float* __restrict__ ald, const float* __restrict__ m,
                            float* __restrict__ z) {
  int e = blockIdx.x * blockDim.x + threadIdx.x;
  if (e >= ETOT) return;
  int s = (e < N_EDGES) ? ei[e] : (e - N_EDGES);
  int d = (e < N_EDGES) ? ei[N_EDGES + e] : (e - N_EDGES);
#pragma unroll
  for (int h = 0; h < HEADS; ++h) {
    float v = leakyf(als[s * HEADS + h] + ald[d * HEADS + h]);
    atomicAdd(&z[d * HEADS + h], expf(v - m[d * HEADS + h]));
  }
}

__global__ void edge_agg(const int* __restrict__ ei, const float* __restrict__ als,
                         const float* __restrict__ ald, const float* __restrict__ m,
                         const float* __restrict__ z, const float* __restrict__ xl,
                         float* __restrict__ agg) {
  long long t = (long long)blockIdx.x * blockDim.x + threadIdx.x;
  if (t >= (long long)ETOT * HID) return;
  int e = (int)(t >> 7);          // HID == 128
  int c = (int)(t & (HID - 1));
  int s = (e < N_EDGES) ? ei[e] : (e - N_EDGES);
  int d = (e < N_EDGES) ? ei[N_EDGES + e] : (e - N_EDGES);
  const float* srcrow = xl + (size_t)s * NC + c;
  __builtin_prefetch(srcrow, 0, 0);                // global_prefetch_b8: pull gather row to GL2
#pragma unroll
  for (int h = 0; h < HEADS; ++h) {
    float v = leakyf(als[s * HEADS + h] + ald[d * HEADS + h]);
    float coeff = expf(v - m[d * HEADS + h]) / (z[d * HEADS + h] + 1e-16f);
    atomicAdd(&agg[(size_t)d * NC + h * HID + c], srcrow[h * HID] * coeff);
  }
}

// ---------------- head-mean + bias + relu + residual ----------------
__global__ void combine_heads(const float* __restrict__ agg, const float* __restrict__ b,
                              const float* __restrict__ hin, float* __restrict__ hout) {
  int t = blockIdx.x * blockDim.x + threadIdx.x;
  if (t >= N_NODES * HID) return;
  int n = t / HID, c = t % HID;
  const float* a = agg + (size_t)n * NC + c;
  float v = (a[0] + a[HID] + a[2 * HID] + a[3 * HID]) * 0.25f + b[c];
  hout[t] = fmaxf(v, 0.f) + hin[t];
}

// ---------------- global mean pool ----------------
__global__ void pool_sum(const float* __restrict__ h, const int* __restrict__ batch,
                         float* __restrict__ sums, float* __restrict__ cnt) {
  int t = blockIdx.x * blockDim.x + threadIdx.x;
  if (t >= N_NODES * HID) return;
  int n = t / HID, c = t % HID;
  int g = batch[n];
  atomicAdd(&sums[g * HID + c], h[t]);
  if (c == 0) atomicAdd(&cnt[g], 1.0f);
}

// ---------------- output MLP, one block ----------------
__global__ __launch_bounds__(256) void mlp_out(const float* __restrict__ sums,
                                               const float* __restrict__ cnt,
                                               const float* __restrict__ W1, const float* __restrict__ b1,
                                               const float* __restrict__ W2, const float* __restrict__ b2,
                                               const float* __restrict__ W3, const float* __restrict__ b3,
                                               float* __restrict__ out) {
  __shared__ float g[NGRAPH * HID];
  __shared__ float t1[NGRAPH * 64];
  __shared__ float t2[NGRAPH * 64];
  int tid = threadIdx.x;
  for (int i = tid; i < NGRAPH * HID; i += 256) {
    int r = i / HID;
    g[i] = sums[i] / fmaxf(cnt[r], 1.0f);
  }
  __syncthreads();
  for (int i = tid; i < NGRAPH * 64; i += 256) {
    int r = i / 64, j = i % 64;
    float acc = b1[j];
    for (int k = 0; k < HID; ++k) acc += g[r * HID + k] * W1[k * 64 + j];
    t1[i] = fmaxf(acc, 0.f);
  }
  __syncthreads();
  for (int i = tid; i < NGRAPH * 64; i += 256) {
    int r = i / 64, j = i % 64;
    float acc = b2[j];
    for (int k = 0; k < 64; ++k) acc += t1[r * 64 + k] * W2[k * 64 + j];
    t2[i] = fmaxf(acc, 0.f);
  }
  __syncthreads();
  for (int i = tid; i < NGRAPH * 32; i += 256) {
    int r = i / 32, j = i % 32;
    float acc = b3[j];
    for (int k = 0; k < 64; ++k) acc += t2[r * 64 + k] * W3[k * 32 + j];
    out[i] = acc;
  }
}

// ---------------- host orchestration ----------------
static inline size_t alignUp(size_t v, size_t a) { return (v + a - 1) & ~(a - 1); }

extern "C" void kernel_launch(void* const* d_in, const int* in_sizes, int n_in,
                              void* d_out, int out_size, void* d_ws, size_t ws_size,
                              hipStream_t stream) {
  (void)in_sizes; (void)n_in; (void)out_size; (void)ws_size;
  const float* x       = (const float*)d_in[0];
  const int*   ei      = (const int*)  d_in[1];
  const int*   batch   = (const int*)  d_in[2];
  const float* W_in    = (const float*)d_in[3];
  const float* b_in    = (const float*)d_in[4];
  const float* gat_W   = (const float*)d_in[5];   // [L,128,512]
  const float* att_src = (const float*)d_in[6];   // [L,4,128]
  const float* att_dst = (const float*)d_in[7];
  const float* gat_b   = (const float*)d_in[8];   // [L,128]
  const float* W1 = (const float*)d_in[9];
  const float* b1 = (const float*)d_in[10];
  const float* W2 = (const float*)d_in[11];
  const float* b2 = (const float*)d_in[12];
  const float* W3 = (const float*)d_in[13];
  const float* b3 = (const float*)d_in[14];
  float* out = (float*)d_out;

  // workspace layout
  char* p = (char*)d_ws;
  size_t off = 0;
  auto take = [&](size_t bytes) { void* r = p + off; off = alignUp(off + bytes, 256); return r; };
  _Float16* hb  = (_Float16*)take((size_t)N_NODES * HID * 2);  // f16 activations
  _Float16* Wt  = (_Float16*)take((size_t)NC * HID * 2);       // f16 transposed layer weight
  float* xl     = (float*)take((size_t)N_NODES * NC * 4);      // h @ W
  float* als    = (float*)take((size_t)N_NODES * HEADS * 4);
  float* ald    = (float*)take((size_t)N_NODES * HEADS * 4);
  float* mbuf   = (float*)take((size_t)N_NODES * HEADS * 4);
  float* zbuf   = (float*)take((size_t)N_NODES * HEADS * 4);
  float* agg    = (float*)take((size_t)N_NODES * NC * 4);
  float* h0     = (float*)take((size_t)N_NODES * HID * 4);
  float* h1     = (float*)take((size_t)N_NODES * HID * 4);
  float* psum   = (float*)take((size_t)NGRAPH * HID * 4);
  float* pcnt   = (float*)take((size_t)NGRAPH * 4);

  const int T = 256;
  const int nNH   = N_NODES * HID;           // 2.56M
  const int bNH   = (nNH + T - 1) / T;
  const int bNHe  = (N_NODES * HEADS + T - 1) / T;
  const int bE    = (ETOT + T - 1) / T;
  const long long nAgg = (long long)ETOT * HID;
  const int bAgg  = (int)((nAgg + T - 1) / T);
  const int bWT   = (NC * HID + T - 1) / T;
  const int gemmBlocks = (N_NODES / 16) * (NC / 64) / 8;  // 16x64 tile per wave, 8 waves/block

  // 1) input projection
  proj_relu<<<bNH, T, 0, stream>>>(x, W_in, b_in, h0);

  float* hcur = h0;
  float* hnext = h1;
  for (int l = 0; l < NLAYERS; ++l) {
    const float* gw = gat_W   + (size_t)l * HID * NC;
    const float* as = att_src + (size_t)l * HEADS * HID;
    const float* ad = att_dst + (size_t)l * HEADS * HID;
    const float* gb = gat_b   + (size_t)l * HID;

    // convert activations + weights to f16
    cvt_f16<<<bNH, T, 0, stream>>>(hcur, hb, nNH);
    wt_f16<<<bWT, T, 0, stream>>>(gw, Wt);

    // xl = hb @ Wt^T via v_wmma_f32_16x16x32_f16 (register-blocked 16x64 per wave)
    gemm_wmma_f16<<<gemmBlocks, T, 0, stream>>>(hb, Wt, xl);

    // attention logits
    att_dots<<<bNHe, T, 0, stream>>>(xl, as, ad, als, ald);

    // init segment buffers
    fill_f32<<<bNHe, T, 0, stream>>>(mbuf, -3.402823466e38f, N_NODES * HEADS);
    fill_f32<<<bNHe, T, 0, stream>>>(zbuf, 0.f, N_NODES * HEADS);
    fill_f32<<<(N_NODES * NC + T - 1) / T, T, 0, stream>>>(agg, 0.f, N_NODES * NC);

    // segment softmax + aggregation over edges (incl. self-loops)
    edge_max   <<<bE,   T, 0, stream>>>(ei, als, ald, mbuf);
    edge_expsum<<<bE,   T, 0, stream>>>(ei, als, ald, mbuf, zbuf);
    edge_agg   <<<bAgg, T, 0, stream>>>(ei, als, ald, mbuf, zbuf, xl, agg);

    // mean over heads + bias + relu + residual
    combine_heads<<<bNH, T, 0, stream>>>(agg, gb, hcur, hnext);

    float* tmp = hcur; hcur = hnext; hnext = tmp;
  }

  // global mean pool
  fill_f32<<<(NGRAPH * HID + T - 1) / T, T, 0, stream>>>(psum, 0.f, NGRAPH * HID);
  fill_f32<<<1, T, 0, stream>>>(pcnt, 0.f, NGRAPH);
  pool_sum<<<bNH, T, 0, stream>>>(hcur, batch, psum, pcnt);

  // output MLP
  mlp_out<<<1, T, 0, stream>>>(psum, pcnt, W1, b1, W2, b2, W3, b3, out);
}
